// Info_Dropout_71949292143338
// MI455X (gfx1250) — matrix operand order repaired and checked by reading, not compiled, not run on hardware
//
#include <hip/hip_runtime.h>
#include <hip/hip_bf16.h>
#include <math.h>

typedef __attribute__((ext_vector_type(16))) _Float16 v16h;
typedef __attribute__((ext_vector_type(8)))  _Float16 v8h;
typedef __attribute__((ext_vector_type(2)))  _Float16 v2h;
typedef __attribute__((ext_vector_type(8)))  float    v8f;

#define B_  32
#define C_  64
#define H_  56
#define W_  56
#define HD  58
#define WD  58
#define NS  9
#define HW  3136        // 56*56
#define N_DIST 903168   // 32*9*56*56
#define N_OUT  6422528  // 32*64*56*56
#define CPAD 72         // channel row padded to 144 B (16B-aligned, conflict-free)

// ---------------------------------------------------------------------------
// Kernel A: dist0[b,k,y,x] = sum_c (center - shifted)^2 on the 58x58 grid.
// One wave32 per (b,k,y, 16-pixel tile). f32 diffs -> f16 in LDS (pixel-major
// so fragments load as ds_load_b128), Gram-matrix trick: D = A x A^T via two
// v_wmma_f32_16x16x32_f16 (K=0..31, K=32..63); diagonal D[p][p] = sum_c d^2.
// ---------------------------------------------------------------------------
__global__ __launch_bounds__(256) void kdist0(const float* __restrict__ xold,
                                              const int* __restrict__ si,
                                              const int* __restrict__ sj,
                                              float* __restrict__ dist0) {
    __shared__ __align__(16) _Float16 sd[8][16][CPAD];  // [wave][pixel][channel]
    const int lane = threadIdx.x & 31;
    const int w    = threadIdx.x >> 5;
    const int g    = blockIdx.x * 8 + w;           // global wave id
    const int tile = g & 3;                        // 4 tiles of 16 px per row
    const int y    = (g >> 2) % HD;
    const int k    = (g / (4 * HD)) % NS;
    const int b    = g / (4 * HD * NS);
    const int di   = si[k] - 3;
    const int dj   = sj[k] - 3;
    const int x0   = tile * 16;
    const int p    = lane & 15;                    // pixel within tile
    const int xp   = x0 + p;
    const int r0 = y - 1,      q0 = xp - 1;        // center in x_old coords
    const int r1 = r0 + di,    q1 = q0 + dj;       // shifted in x_old coords
    const bool okC = (unsigned)r0 < (unsigned)H_ && (unsigned)q0 < (unsigned)W_;
    const bool okS = (unsigned)r1 < (unsigned)H_ && (unsigned)q1 < (unsigned)W_;
    const long baseB = (long)b * C_ * H_ * W_;

    // Each lane covers pixel p; two adjacent channels per iteration, packed
    // into one dword LDS store (conflict-free: stride 36 banks across lanes).
    #pragma unroll 4
    for (int it = 0; it < 16; ++it) {
        int c0 = it * 4 + ((lane >> 4) << 1);      // {0,4,..} or {2,6,..}
        const float* cp0 = xold + baseB + (long)c0 * (H_ * W_);
        const float* cp1 = cp0 + (H_ * W_);
        float cv0 = 0.f, sv0 = 0.f, cv1 = 0.f, sv1 = 0.f;
        if (okC) { cv0 = cp0[r0 * W_ + q0]; cv1 = cp1[r0 * W_ + q0]; }
        if (okS) { sv0 = cp0[r1 * W_ + q1]; sv1 = cp1[r1 * W_ + q1]; }
        v2h d2 = { (_Float16)(cv0 - sv0), (_Float16)(cv1 - sv1) };
        *(v2h*)&sd[w][p][c0] = d2;
    }
    asm volatile("" ::: "memory");   // keep LDS store->load order (same-wave LDS is in-order)

    // Build WMMA fragments in contiguous K order (same for A and B), as b128 loads.
    const int m   = lane & 15;
    const int hiA = (lane >> 4) << 3;   // unused: A now uses the same K order as B (kb)
    const int kb  = (lane >> 4) << 4;   // B: lanes 0-15 -> K 0..15, 16-31 -> K 16..31
    const _Float16* row = &sd[w][m][0];
    v8h aA = *(const v8h*)(row + kb);
    v8h aB = *(const v8h*)(row + kb + 8);
    v8h aC = *(const v8h*)(row + 32 + kb);
    v8h aD = *(const v8h*)(row + 32 + kb + 8);
    v8h bA = *(const v8h*)(row + kb);
    v8h bB = *(const v8h*)(row + kb + 8);
    v8h bC = *(const v8h*)(row + 32 + kb);
    v8h bD = *(const v8h*)(row + 32 + kb + 8);
    v16h a0 = __builtin_shufflevector(aA, aB, 0,1,2,3,4,5,6,7,8,9,10,11,12,13,14,15);
    v16h a1 = __builtin_shufflevector(aC, aD, 0,1,2,3,4,5,6,7,8,9,10,11,12,13,14,15);
    v16h b0 = __builtin_shufflevector(bA, bB, 0,1,2,3,4,5,6,7,8,9,10,11,12,13,14,15);
    v16h b1 = __builtin_shufflevector(bC, bD, 0,1,2,3,4,5,6,7,8,9,10,11,12,13,14,15);

    v8f acc = {};
    acc = __builtin_amdgcn_wmma_f32_16x16x32_f16(false, a0, false, b0, (short)0, acc, false, false);
    acc = __builtin_amdgcn_wmma_f32_16x16x32_f16(false, a1, false, b1, (short)0, acc, false, false);

    float tmp[8];
    #pragma unroll
    for (int j = 0; j < 8; ++j) tmp[j] = acc[j];
    // Diagonal of C/D layout: p<8 -> lane p, vgpr p ; p>=8 -> lane 16+p, vgpr p-8.
    int   pd = -1;
    float v  = 0.f;
    if (lane < 8)        { pd = lane;      v = tmp[lane]; }
    else if (lane >= 24) { pd = lane - 16; v = tmp[lane - 24]; }
    if (pd >= 0) {
        int xo = x0 + pd;
        if (xo < WD) dist0[(((long)b * NS + k) * HD + y) * WD + xo] = v;
    }
}

// ---------------------------------------------------------------------------
// Kernel B: 3x3 VALID box sum -> dist[b,k,56,56]; per-block partial sums
// (deterministic tree reduction) for the global mean.
// ---------------------------------------------------------------------------
__global__ __launch_bounds__(256) void kwin(const float* __restrict__ dist0,
                                            float* __restrict__ dist,
                                            float* __restrict__ part) {
    __shared__ float red[256];
    const int idx = blockIdx.x * 256 + threadIdx.x;   // grid sized exactly
    const int x  = idx % W_;
    const int y  = (idx / W_) % H_;
    const int bk = idx / HW;
    const float* src = dist0 + (long)bk * (HD * WD) + y * WD + x;
    float s = 0.f;
    #pragma unroll
    for (int u = 0; u < 3; ++u)
        #pragma unroll
        for (int vv = 0; vv < 3; ++vv)
            s += src[u * WD + vv];
    dist[idx] = s;
    red[threadIdx.x] = s;
    __syncthreads();
    for (int st = 128; st > 0; st >>= 1) {
        if (threadIdx.x < st) red[threadIdx.x] += red[threadIdx.x + st];
        __syncthreads();
    }
    if (threadIdx.x == 0) part[blockIdx.x] = red[0];
}

// ---------------------------------------------------------------------------
// Kernel B2: reduce 3528 partials (fixed order -> deterministic), publish
// invDenom = 1/(2*mean) so kernel C computes exp(-d * invDenom).
// ---------------------------------------------------------------------------
__global__ __launch_bounds__(256) void kmean(const float* __restrict__ part,
                                             float* __restrict__ scal) {
    __shared__ float red[256];
    float s = 0.f;
    for (int i = threadIdx.x; i < 3528; i += 256) s += part[i];
    red[threadIdx.x] = s;
    __syncthreads();
    for (int st = 128; st > 0; st >>= 1) {
        if (threadIdx.x < st) red[threadIdx.x] += red[threadIdx.x + st];
        __syncthreads();
    }
    if (threadIdx.x == 0) {
        float mean = red[0] / (float)N_DIST;
        scal[0] = 1.0f / (mean * 2.0f);   // BAND_WIDTH = 1
    }
}

// ---------------------------------------------------------------------------
// Kernel C: one block per batch. pu = (mean_k exp(-d*inv))^(1/T) kept in LDS,
// deterministic per-batch spatial sum, then factor map
// fmap[b,pos] = exp(-1.5 * (pu/S_b) * 3136) written to global.
// ---------------------------------------------------------------------------
__global__ __launch_bounds__(256) void kprob(const float* __restrict__ dist,
                                             const float* __restrict__ scal,
                                             float* __restrict__ fmap) {
    __shared__ float spu[HW];     // 12.25 KB
    __shared__ float red[256];
    const int b = blockIdx.x;
    const float inv = scal[0];
    const float pexp = (float)(1.0 / 0.03);   // 33.3333...
    float ls = 0.f;
    for (int pos = threadIdx.x; pos < HW; pos += 256) {
        float s = 0.f;
        #pragma unroll
        for (int k = 0; k < NS; ++k)
            s += expf(-dist[((long)b * NS + k) * HW + pos] * inv);
        float pu = powf(s * (1.0f / 9.0f), pexp);
        spu[pos] = pu;
        ls += pu;
    }
    red[threadIdx.x] = ls;
    __syncthreads();
    for (int st = 128; st > 0; st >>= 1) {
        if (threadIdx.x < st) red[threadIdx.x] += red[threadIdx.x + st];
        __syncthreads();
    }
    const float S = red[0];
    for (int pos = threadIdx.x; pos < HW; pos += 256)
        fmap[(long)b * HW + pos] = expf(-1.5f * (spu[pos] / S) * (float)HW);
}

// ---------------------------------------------------------------------------
// Kernel D: out = x * fmap (broadcast over channels). Pure bandwidth.
// ---------------------------------------------------------------------------
__global__ __launch_bounds__(256) void kapply(const float* __restrict__ x,
                                              const float* __restrict__ fmap,
                                              float* __restrict__ out) {
    const long idx = (long)blockIdx.x * 256 + threadIdx.x;
    const int pos = (int)(idx % HW);
    const int b   = (int)(idx / ((long)HW * C_));
    __builtin_prefetch(x + idx + 8192, 0, 0);   // global_prefetch_b8
    out[idx] = x[idx] * fmap[(long)b * HW + pos];
}

extern "C" void kernel_launch(void* const* d_in, const int* in_sizes, int n_in,
                              void* d_out, int out_size, void* d_ws, size_t ws_size,
                              hipStream_t stream) {
    const float* xold = (const float*)d_in[0];
    const float* x    = (const float*)d_in[1];
    const int*   si   = (const int*)d_in[2];
    const int*   sj   = (const int*)d_in[3];
    float* out = (float*)d_out;
    float* ws  = (float*)d_ws;

    // workspace layout (floats): ~7.9 MB total
    float* dist0 = ws;                         // 32*9*58*58 = 968832
    float* dist  = dist0 + 968832;             // 32*9*56*56 = 903168
    float* part  = dist  + N_DIST;             // 3528
    float* scal  = part  + 3528;               // 8 (padded)
    float* fmap  = scal  + 8;                  // 32*3136 = 100352

    kdist0<<<8352, 256, 0, stream>>>(xold, si, sj, dist0);   // 66816 waves exact
    kwin  <<<3528, 256, 0, stream>>>(dist0, dist, part);     // 903168 threads exact
    kmean <<<1,    256, 0, stream>>>(part, scal);
    kprob <<<32,   256, 0, stream>>>(dist, scal, fmap);
    kapply<<<25088,256, 0, stream>>>(x, fmap, out);          // 6422528 threads exact
}
